// Head_26628797235819
// MI455X (gfx1250) — matrix operand order repaired and checked by reading, not compiled, not run on hardware
//
#include <hip/hip_runtime.h>
#include <hip/hip_bf16.h>

typedef __attribute__((ext_vector_type(16))) _Float16 v16h;
typedef __attribute__((ext_vector_type(8)))  float    v8f;
typedef __attribute__((ext_vector_type(4)))  unsigned int u32x4;

#define B_  4
#define T_  4096
#define E_  1024
#define H_  64
#define BT_ (B_*T_)

union AFrag { v16h v; u32x4 q[2]; unsigned int u[8]; };
union H4    { _Float16 f[4]; unsigned long long ull; };

__device__ inline unsigned pkh(float a, float b) {   // exactly one v_cvt_pk
  auto r = __builtin_amdgcn_cvt_pkrtz(a, b);         // __fp16 ext_vector(2)
  return __builtin_bit_cast(unsigned, r);
}

// CDNA5 async DMA: global -> LDS, 16B per lane, tracked by ASYNCcnt.
__device__ inline void async_load_b128(void* lds_ptr, const void* gptr) {
  unsigned lds_off = (unsigned)(size_t)lds_ptr;      // LDS flat addr low 32 = offset
  asm volatile("global_load_async_to_lds_b128 %0, %1, off"
               :: "v"(lds_off), "v"(gptr) : "memory");
}

// ---------------------------------------------------------------------------
// Kernel 0: one-time weight prep.  W[1024,64] f32 (q,k,v) -> WhT[3][64][1024]
// f16 transposed; 384 KB, L2-resident for the whole dispatch.
// ---------------------------------------------------------------------------
__global__ __launch_bounds__(256) void wprep_kernel(
    const float* __restrict__ Wk, const float* __restrict__ Wq,
    const float* __restrict__ Wv, _Float16* __restrict__ WhT)
{
  int gid = blockIdx.x * 256 + threadIdx.x;
  int p   = gid >> 14;
  int rem = gid & 16383;
  int c   = rem >> 8;
  int k4  = (rem & 255) << 2;
  const float* W = (p == 0) ? Wq : (p == 1) ? Wk : Wv;
  H4 h;
  #pragma unroll
  for (int i = 0; i < 4; ++i)
    h.f[i] = (_Float16)W[(size_t)(k4 + i) * H_ + c];
  *(unsigned long long*)&WhT[(size_t)p * 64 * E_ + (size_t)c * E_ + k4] = h.ull;
}

// ---------------------------------------------------------------------------
// Kernel 1: q/k/v projections, barrier-free hot loop.  A fragments straight
// from x (f32 -> cvt_pkrtz), B fragments from L2-resident WhT.
// Epilogue: q/k tiles bounce through 4 KB LDS so stores are b128.
// ---------------------------------------------------------------------------
__global__ __launch_bounds__(384) void qkv_proj_kernel(
    const float* __restrict__ x, const _Float16* __restrict__ WhT,
    _Float16* __restrict__ qh, _Float16* __restrict__ kh,
    _Float16* __restrict__ vT)
{
  __shared__ _Float16 st[8][16][16];         // q/k epilogue transpose (waves 0-7)

  const int tid  = threadIdx.x;
  const int wave = tid >> 5, lane = tid & 31;
  const int hiL  = lane >> 4, lo = lane & 15;
  const int proj = wave >> 2;                // 0=q, 1=k, 2=v
  const int c0   = (wave & 3) << 4;
  const int m0   = blockIdx.x << 4;

  const float*    xrow = x + (size_t)(m0 + lo) * E_;
  const _Float16* wrow = WhT + (size_t)proj * 64 * E_ + (size_t)(c0 + lo) * E_;

  v8f acc = {};

  for (int k0 = 0; k0 < E_; k0 += 32) {
    float4 fa0 = *(const float4*)&xrow[k0 + (hiL << 3)];
    float4 fa1 = *(const float4*)&xrow[k0 + (hiL << 3) + 4];
    float4 fa2 = *(const float4*)&xrow[k0 + 16 + (hiL << 3)];
    float4 fa3 = *(const float4*)&xrow[k0 + 16 + (hiL << 3) + 4];
    AFrag a;
    a.u[0] = pkh(fa0.x, fa0.y); a.u[1] = pkh(fa0.z, fa0.w);
    a.u[2] = pkh(fa1.x, fa1.y); a.u[3] = pkh(fa1.z, fa1.w);
    a.u[4] = pkh(fa2.x, fa2.y); a.u[5] = pkh(fa2.z, fa2.w);
    a.u[6] = pkh(fa3.x, fa3.y); a.u[7] = pkh(fa3.z, fa3.w);

    AFrag bf;
    bf.q[0] = *(const u32x4*)&wrow[k0 + (hiL << 3)];
    bf.q[1] = *(const u32x4*)&wrow[k0 + 16 + (hiL << 3)];

    acc = __builtin_amdgcn_wmma_f32_16x16x32_f16(false, a.v, false, bf.v,
                                                 (short)0, acc, false, false);
  }

  if (proj < 2) {                            // q, k: row-major via LDS transpose
    _Float16* dst = (proj == 0) ? qh : kh;
    #pragma unroll
    for (int j = 0; j < 8; ++j) {
      int M = j + (hiL << 3);
      st[wave][M][lo] = (_Float16)acc[j];
    }
    asm volatile("s_wait_dscnt 0" ::: "memory");   // wave-private LDS RAW
    u32x4 rowv = *(const u32x4*)&st[wave][lo][hiL << 3];
    *(u32x4*)&dst[(size_t)(m0 + lo) * H_ + c0 + (hiL << 3)] = rowv;
  } else {                                   // v: transposed vT[64][BT] (contig)
    #pragma unroll
    for (int j = 0; j < 8; ++j) {
      int M = j + (hiL << 3);
      vT[(size_t)(c0 + lo) * BT_ + m0 + M] = (_Float16)acc[j];
    }
  }
}

// ---------------------------------------------------------------------------
// Kernel 2: flash attention.  8 waves/block, 16 query rows/wave.  K/V chunks
// double-buffered in LDS via CDNA5 async DMA; DMA of chunk c+1 overlaps the
// WMMA + softmax of chunk c.
// ---------------------------------------------------------------------------
__global__ __launch_bounds__(256) void flash_attn_kernel(
    const _Float16* __restrict__ qh, const _Float16* __restrict__ kh,
    const _Float16* __restrict__ vT, float* __restrict__ out)
{
  __shared__ _Float16 ks[2][32][64];   // K chunks, row-major
  __shared__ _Float16 vsT[2][64][32];  // V chunks (pre-transposed in global)
  __shared__ _Float16 ps[8][16][32];   // per-wave P staging (C->A relayout)

  const int tid   = threadIdx.x;
  const int wave  = tid >> 5, lane = tid & 31;
  const int hiL   = lane >> 4, lo = lane & 15;
  const int b     = blockIdx.y;
  const int qbase = blockIdx.x << 7;
  const int q0    = qbase + (wave << 4);
  const size_t base  = (size_t)b * T_ * H_;
  const size_t vbase = (size_t)b * T_;

  const int skey = tid >> 3, sh8 = (tid & 7) << 3;    // K slot: [key][h8..+7]
  const int vcol = tid >> 2, vp8 = (tid & 3) << 3;    // V slot: [col][p8..+7]

  AFrag aq[2];                                         // resident Q fragments
  #pragma unroll
  for (int t = 0; t < 2; ++t) {
    const _Float16* qrow = &qh[base + (size_t)(q0 + lo) * H_ + (t << 5)];
    aq[t].q[0] = *(const u32x4*)&qrow[hiL << 3];
    aq[t].q[1] = *(const u32x4*)&qrow[16 + (hiL << 3)];
  }

  v8f o[4] = {{}, {}, {}, {}};
  float mrow[8], lrow[8];
  #pragma unroll
  for (int j = 0; j < 8; ++j) { mrow[j] = -1e30f; lrow[j] = 0.f; }

  const int nchunks = (qbase >> 5) + 4;

  // prime buffer 0 with chunk 0
  async_load_b128(&ks[0][skey][sh8], &kh[base + (size_t)skey * H_ + sh8]);
  async_load_b128(&vsT[0][vcol][vp8], &vT[(size_t)vcol * BT_ + vbase + vp8]);

  for (int c = 0; c < nchunks; ++c) {
    const int cur    = c & 1;
    const int kstart = c << 5;
    __syncthreads();                         // buf[cur^1] free for refill

    const int cn = (c + 1 < nchunks) ? c + 1 : c;        // clamped next chunk
    const int kn = cn << 5;
    async_load_b128(&ks[cur ^ 1][skey][sh8],
                    &kh[base + (size_t)(kn + skey) * H_ + sh8]);
    async_load_b128(&vsT[cur ^ 1][vcol][vp8],
                    &vT[(size_t)vcol * BT_ + vbase + kn + vp8]);
    const int cp = (c + 2 < nchunks) ? c + 2 : c;        // warm L2 two ahead
    __builtin_prefetch(&kh[base + (size_t)((cp << 5) + skey) * H_ + sh8], 0, 1);
    __builtin_prefetch(&vT[(size_t)vcol * BT_ + vbase + (cp << 5) + vp8], 0, 1);

    asm volatile("s_wait_asynccnt 2" ::: "memory");      // my buf[cur] landed
    __syncthreads();                                      // everyone's landed

    if (kstart <= q0 + 15) {                 // wave-uniform branch
      v8f s[2];
      #pragma unroll
      for (int sub = 0; sub < 2; ++sub) {
        v8f sacc = {};
        #pragma unroll
        for (int t = 0; t < 2; ++t) {
          AFrag bk;                          // B = K^T
          const _Float16* krow = &ks[cur][(sub << 4) + lo][(t << 5) + (hiL << 3)];
          bk.q[0] = *(const u32x4*)&krow[0];
          bk.q[1] = *(const u32x4*)&krow[16];
          sacc = __builtin_amdgcn_wmma_f32_16x16x32_f16(false, aq[t].v, false, bk.v,
                                                        (short)0, sacc, false, false);
        }
        #pragma unroll
        for (int j = 0; j < 8; ++j) {        // scale + causal mask
          int qg  = q0 + j + (hiL << 3);
          int key = kstart + (sub << 4) + lo;
          sacc[j] = (key <= qg) ? sacc[j] * 0.125f : -1e30f;
        }
        s[sub] = sacc;
      }

      #pragma unroll
      for (int j = 0; j < 8; ++j) {          // online softmax per row
        float mx = fmaxf(s[0][j], s[1][j]);
        #pragma unroll
        for (int d = 1; d < 16; d <<= 1) mx = fmaxf(mx, __shfl_xor(mx, d, 32));
        float mnew  = fmaxf(mrow[j], mx);
        float alpha = __expf(mrow[j] - mnew);
        float p0 = __expf(s[0][j] - mnew);
        float p1 = __expf(s[1][j] - mnew);
        float rs = p0 + p1;
        #pragma unroll
        for (int d = 1; d < 16; d <<= 1) rs += __shfl_xor(rs, d, 32);
        lrow[j] = lrow[j] * alpha + rs;
        mrow[j] = mnew;
        #pragma unroll
        for (int nt = 0; nt < 4; ++nt) o[nt][j] *= alpha;
        int M = j + (hiL << 3);
        ps[wave][M][lo]      = (_Float16)p0;
        ps[wave][M][16 + lo] = (_Float16)p1;
      }
      asm volatile("s_wait_dscnt 0" ::: "memory");       // wave-private LDS RAW

      AFrag pa;
      pa.q[0] = *(const u32x4*)&ps[wave][lo][hiL << 3];
      pa.q[1] = *(const u32x4*)&ps[wave][lo][16 + (hiL << 3)];
      #pragma unroll
      for (int nt = 0; nt < 4; ++nt) {
        AFrag bv;                            // B = V
        const _Float16* vrow = &vsT[cur][(nt << 4) + lo][hiL << 3];
        bv.q[0] = *(const u32x4*)&vrow[0];
        bv.q[1] = *(const u32x4*)&vrow[16];
        o[nt] = __builtin_amdgcn_wmma_f32_16x16x32_f16(false, pa.v, false, bv.v,
                                                       (short)0, o[nt], false, false);
      }
    }
  }

  #pragma unroll
  for (int nt = 0; nt < 4; ++nt)
    #pragma unroll
    for (int j = 0; j < 8; ++j) {
      int M = j + (hiL << 3);
      out[base + (size_t)(q0 + M) * H_ + (nt << 4) + lo] = o[nt][j] / lrow[j];
    }
}

// ---------------------------------------------------------------------------
extern "C" void kernel_launch(void* const* d_in, const int* in_sizes, int n_in,
                              void* d_out, int out_size, void* d_ws, size_t ws_size,
                              hipStream_t stream) {
  const float* x  = (const float*)d_in[0];
  const float* Wk = (const float*)d_in[1];
  const float* Wq = (const float*)d_in[2];
  const float* Wv = (const float*)d_in[3];
  float* out = (float*)d_out;

  _Float16* qh  = (_Float16*)d_ws;                    // [BT][64]
  _Float16* kh  = qh + (size_t)BT_ * H_;              // [BT][64]
  _Float16* vT  = kh + (size_t)BT_ * H_;              // [64][BT]
  _Float16* WhT = vT + (size_t)BT_ * H_;              // [3][64][1024]

  wprep_kernel<<<dim3(192), dim3(256), 0, stream>>>(Wk, Wq, Wv, WhT);
  qkv_proj_kernel<<<dim3(BT_ / 16), dim3(384), 0, stream>>>(x, WhT, qh, kh, vT);
  flash_attn_kernel<<<dim3(T_ / 128, B_), dim3(256), 0, stream>>>(qh, kh, vT, out);
}
